// GCNModule_83348135346604
// MI455X (gfx1250) — matrix operand order repaired and checked by reading, not compiled, not run on hardware
//
#include <hip/hip_runtime.h>
#include <hip/hip_bf16.h>

typedef __attribute__((ext_vector_type(16))) _Float16 v16h;
typedef __attribute__((ext_vector_type(8)))  float    v8f;

#define S_   512
#define P_   128
#define G_   16
#define DIN  256
#define DHID 512
#define DOUT 16
#define DFIN 64

// Repack W_intra_0 (DIN x DHID, f32 row-major) into f16 WMMA B-fragment layout:
// Bpack[((kt*DHID + n)*32) + ki]  where k = kt*32 + ki  (32 contiguous K-halfs per column)
__global__ void pack_w0_kernel(const float* __restrict__ W0, _Float16* __restrict__ Bpack) {
    int idx = blockIdx.x * blockDim.x + threadIdx.x;        // over DIN*DHID
    if (idx >= DIN * DHID) return;
    int k = idx / DHID;
    int n = idx - k * DHID;
    int kt = k >> 5;
    int ki = k & 31;
    Bpack[((kt * DHID + n) << 5) + ki] = (_Float16)W0[idx];
}

__global__ void __launch_bounds__(256)
gcn_scene_kernel(const float* __restrict__ h_states,
                 const int*   __restrict__ end_group,
                 const _Float16* __restrict__ Bpack,   // packed W_intra_0 (f16)
                 const float* __restrict__ W1,         // W_intra_1 (512x16)
                 const float* __restrict__ Wi0,        // W_inter_0 (16x512)
                 const float* __restrict__ Wi1,        // W_inter_1 (512x16)
                 const float* __restrict__ outW,       // (32x64)
                 const float* __restrict__ outB,       // (64)
                 float* __restrict__ out) {
    const int scene = blockIdx.x;
    const int t = threadIdx.x;                 // 256 threads = 8 wave32
    const int lane = t & 31;
    const int wave = t >> 5;

    __shared__ int      sLab[P_];
    __shared__ int      sCnt[G_];
    __shared__ float    sGh [G_ * DIN];        // 16 KB  group sums/means (f32)
    __shared__ _Float16 sGhH[G_ * DIN];        //  8 KB  group means (f16, A operand)
    __shared__ float    sHg [G_ * DHID];       // 32 KB  relu(Gh @ W0)
    __shared__ float    sW1 [DHID * DOUT];     // 32 KB  staged W_intra_1
    __shared__ float    sHi [G_ * DOUT];       // Hi_g
    __shared__ float    sMv [DOUT];            // group-mean vector
    __shared__ float    sTv [DHID];            // relu(m @ Wi0)
    __shared__ float    sV  [DOUT];            // relu(t @ Wi1)/8
    __shared__ float    sW2 [DFIN];            // inter-term + bias (shared by all rows)
    __shared__ float    sOut[G_ * DFIN];       // 16 distinct output rows

    const float* hS = h_states + (size_t)scene * P_ * DIN;

    // ---- Phase A: labels, counts, zero accumulators -------------------------
    if (t < P_)  sLab[t] = end_group[scene * P_ + t];
    if (t < G_)  sCnt[t] = 0;
    #pragma unroll
    for (int i = 0; i < G_; ++i) sGh[i * DIN + t] = 0.0f;
    __syncthreads();
    if (t < P_) atomicAdd(&sCnt[sLab[t] - 1], 1);
    __syncthreads();

    // ---- Phase B: group sums (thread t owns column t -> no write conflicts) --
    for (int p = 0; p < P_; ++p)
        sGh[(sLab[p] - 1) * DIN + t] += hS[p * DIN + t];
    __syncthreads();

    // means -> f16 A operand; also stage W_intra_1 for the tail GEMM
    #pragma unroll
    for (int g = 0; g < G_; ++g)
        sGhH[g * DIN + t] = (_Float16)(sGh[g * DIN + t] / (float)sCnt[g]);
    for (int i = t; i < DHID * DOUT; i += 256) sW1[i] = W1[i];
    __syncthreads();

    // ---- Phase C: GEMM1 via WMMA: sHg = relu(Gh(16x256) @ W0(256x512)) -------
    {
        const int mRow = lane & 15;
        const int aK   = (lane >> 4) * 16;     // lanes 0-15: K 0-15 ; 16-31: K 16-31
        v16h afrag[8];
        #pragma unroll
        for (int kt = 0; kt < 8; ++kt) {
            const _Float16* ap = &sGhH[mRow * DIN + kt * 32 + aK];
            #pragma unroll
            for (int i = 0; i < 8; ++i) {
                afrag[kt][i]     = ap[i];       // K = base+0..7
                afrag[kt][8 + i] = ap[8 + i];   // K = base+8..15
            }
        }
        const int nOff = lane & 15;
        const int bK   = (lane >> 4) * 16;     // lanes 0-15: K 0-15 ; lanes 16-31: K 16-31
        #pragma unroll
        for (int j = 0; j < 4; ++j) {
            const int nt = wave * 4 + j;       // 8 waves x 4 = 32 N-tiles of 16
            const int n  = nt * 16 + nOff;
            v8f c = {};
            #pragma unroll
            for (int kt = 0; kt < 8; ++kt) {
                const v16h b = *(const v16h*)(Bpack + (((size_t)(kt * DHID + n)) << 5) + bK);
                c = __builtin_amdgcn_wmma_f32_16x16x32_f16(
                        false, afrag[kt], false, b, (short)0, c, false, false);
            }
            const int m0 = (lane >> 4) * 8;    // C rows: lanes<16 -> M r, lanes>=16 -> M 8+r
            #pragma unroll
            for (int r = 0; r < 8; ++r)
                sHg[(m0 + r) * DHID + nt * 16 + nOff] = fmaxf(c[r], 0.0f);
        }
    }
    __syncthreads();

    // ---- Phase D: Hi_g = relu(Hg(16x512) @ W1(512x16)) -----------------------
    {
        const int m = t >> 4, o = t & 15;
        float acc = 0.0f;
        for (int k = 0; k < DHID; ++k)
            acc = fmaf(sHg[m * DHID + k], sW1[k * DOUT + o], acc);
        sHi[m * DOUT + o] = fmaxf(acc, 0.0f);
    }
    __syncthreads();

    // ---- Phase E: inter-group path (rows of Ai@X are identical) --------------
    if (t < DOUT) {                            // column means of Hi_g
        float acc = 0.0f;
        #pragma unroll
        for (int m = 0; m < G_; ++m) acc += sHi[m * DOUT + t];
        sMv[t] = acc * (1.0f / (float)G_);
    }
    __syncthreads();
    {                                          // t_vec = relu(m @ Wi0), 512 wide
        #pragma unroll
        for (int h = 0; h < 2; ++h) {
            const int o = t + h * 256;
            float acc = 0.0f;
            #pragma unroll
            for (int k = 0; k < DOUT; ++k)
                acc = fmaf(sMv[k], Wi0[k * DHID + o], acc);
            sTv[o] = fmaxf(acc, 0.0f);
        }
    }
    __syncthreads();
    if (t < DOUT) {                            // v = relu(t_vec @ Wi1); fold GUnpool /8
        float acc = 0.0f;
        for (int k = 0; k < DHID; ++k)
            acc = fmaf(sTv[k], Wi1[k * DOUT + t], acc);
        sV[t] = fmaxf(acc, 0.0f) * 0.125f;
    }
    __syncthreads();

    // ---- Phase F: output rows: out_g = Hi_g@outW[:16] + v@outW[16:] + b ------
    if (t < DFIN) {
        float acc = outB[t];
        #pragma unroll
        for (int k = 0; k < DOUT; ++k)
            acc = fmaf(sV[k], outW[(G_ + k) * DFIN + t], acc);
        sW2[t] = acc;
    }
    __syncthreads();
    {
        const int m = t >> 4, o0 = (t & 15) * 4;
        #pragma unroll
        for (int j = 0; j < 4; ++j) {
            const int o = o0 + j;
            float acc = sW2[o];
            #pragma unroll
            for (int k = 0; k < DOUT; ++k)
                acc = fmaf(sHi[m * DOUT + k], outW[k * DFIN + o], acc);
            sOut[m * DFIN + o] = acc;
        }
    }
    __syncthreads();

    // ---- Phase G: coalesced scatter of 16 rows to 128 pedestrians ------------
    float* oS = out + (size_t)scene * P_ * DFIN;
    #pragma unroll 4
    for (int it = 0; it < (P_ * DFIN) / 256; ++it) {
        const int e = it * 256 + t;
        const int p = e >> 6;
        const int c = e & 63;
        oS[e] = sOut[(sLab[p] - 1) * DFIN + c];
    }
}

extern "C" void kernel_launch(void* const* d_in, const int* in_sizes, int n_in,
                              void* d_out, int out_size, void* d_ws, size_t ws_size,
                              hipStream_t stream) {
    const float* h_states = (const float*)d_in[0];
    // d_in[1] end_pos unused by the reference
    // d_in[2] seq_start_end: scenes are contiguous blocks of P_=128
    const int*   end_group = (const int*)d_in[3];
    const float* W0   = (const float*)d_in[4];
    const float* W1   = (const float*)d_in[5];
    const float* Wi0  = (const float*)d_in[6];
    const float* Wi1  = (const float*)d_in[7];
    const float* outW = (const float*)d_in[8];
    const float* outB = (const float*)d_in[9];
    float* out = (float*)d_out;

    _Float16* Bpack = (_Float16*)d_ws;   // DIN*DHID halfs = 256 KB

    pack_w0_kernel<<<(DIN * DHID + 255) / 256, 256, 0, stream>>>(W0, Bpack);
    gcn_scene_kernel<<<S_, 256, 0, stream>>>(h_states, end_group, Bpack,
                                             W1, Wi0, Wi1, outW, outB, out);
}
